// GroupedQueryAttention_55834574848032
// MI455X (gfx1250) — matrix operand order repaired and checked by reading, not compiled, not run on hardware
//
#include <hip/hip_runtime.h>
#include <hip/hip_bf16.h>

typedef __attribute__((ext_vector_type(16))) __bf16 v16bf;
typedef __attribute__((ext_vector_type(4)))  __bf16 v4bf;
typedef __attribute__((ext_vector_type(8)))  float  v8f;

#define B_DIM  4
#define T_DIM  2048
#define E_DIM  2048
#define G_DIM  4
#define QG_DIM 4
#define D_DIM  128
#define HQ_DIM 16                       // G*QG
#define NTOK   (B_DIM * T_DIM)          // 8192

// ---------------------------------------------------------------------------
// CDNA5 async global->LDS copy (ASYNCcnt path), guarded so the code degrades
// to the register-staged pipeline if the builtin is unavailable.
// Builtin signature (from compiler diagnostic): first param is
// "int __attribute__((vector_size(16))) __device__ *" i.e. int4 addrspace(1)*.
// ---------------------------------------------------------------------------
#if __has_builtin(__builtin_amdgcn_global_load_async_to_lds_b128)
#define HAS_ASYNC_LDS 1
#else
#define HAS_ASYNC_LDS 0
#endif

typedef int v4i_vs __attribute__((vector_size(16)));
typedef __attribute__((address_space(1))) v4i_vs as1_v4i;
typedef __attribute__((address_space(3))) v4i_vs as3_v4i;

__device__ __forceinline__ void async_b128(const void* g, void* l) {
#if HAS_ASYNC_LDS
    __builtin_amdgcn_global_load_async_to_lds_b128((as1_v4i*)g, (as3_v4i*)l, 0, 0);
#else
    (void)g; (void)l;
#endif
}

__device__ __forceinline__ void wait_async0() {
#if __has_builtin(__builtin_amdgcn_s_wait_asynccnt)
    __builtin_amdgcn_s_wait_asynccnt(0);
#elif HAS_ASYNC_LDS
    asm volatile("s_wait_asynccnt 0x0" ::: "memory");
#endif
}

// ---------------------------------------------------------------------------
// fp32 -> bf16 conversion (vectorized x4), no transpose (activation side)
// ---------------------------------------------------------------------------
__global__ __launch_bounds__(256) void cvt_f32_bf16(const float* __restrict__ in,
                                                    __bf16* __restrict__ out, int n4) {
    int i = blockIdx.x * 256 + threadIdx.x;
    if (i < n4) {
        float4 v = ((const float4*)in)[i];
        v4bf o;
        o[0] = (__bf16)v.x; o[1] = (__bf16)v.y; o[2] = (__bf16)v.z; o[3] = (__bf16)v.w;
        ((v4bf*)out)[i] = o;
    }
}

// ---------------------------------------------------------------------------
// fp32 [K][N] -> bf16 transposed [N][K]  (weights: makes WMMA B-fragments
// per-lane k-contiguous -> ds_load_b128 instead of 16 scalar ds loads)
// ---------------------------------------------------------------------------
__global__ __launch_bounds__(256) void cvt_transpose_bf16(const float* __restrict__ in,
                                                          __bf16* __restrict__ out,
                                                          int K, int N) {
    __shared__ __bf16 t[64][72];
    const int k0 = blockIdx.y * 64;
    const int n0 = blockIdx.x * 64;
    const int r  = threadIdx.x >> 2;        // 0..63
    const int cb = (threadIdx.x & 3) * 16;  // 0,16,32,48
    const float* src = in + (size_t)(k0 + r) * N + n0 + cb;
#pragma unroll
    for (int u = 0; u < 16; ++u)
        t[cb + u][r] = (__bf16)src[u];
    __syncthreads();
    __bf16* dst = out + (size_t)(n0 + r) * K + k0 + cb;
#pragma unroll
    for (int u = 0; u < 16; ++u)
        dst[u] = t[r][cb + u];
}

// ---------------------------------------------------------------------------
// In-place RoPE on bf16 buffer laid out [B][T][H][D], half = 64
// ---------------------------------------------------------------------------
__global__ __launch_bounds__(256) void rope_kernel(__bf16* __restrict__ q, int H) {
    size_t idx = (size_t)blockIdx.x * 256 + threadIdx.x;
    int    j   = (int)(idx & 63);
    size_t rh  = idx >> 6;                                 // (b*T + t)*H + h
    int    t   = (int)((rh / (size_t)H) % T_DIM);
    __bf16* p  = q + rh * D_DIM;
    float x1 = (float)p[j];
    float x2 = (float)p[j + 64];
    float ang = (float)t * __powf(10000.0f, -(float)j * (1.0f / 64.0f));
    float s, c;
    __sincosf(ang, &s, &c);
    p[j]      = (__bf16)(x1 * c - x2 * s);
    p[j + 64] = (__bf16)(x1 * s + x2 * c);
}

// ---------------------------------------------------------------------------
// bf16 GEMM: C[M,N] = A[M,K] @ W[K,N], with W supplied TRANSPOSED (Bt[N][K]).
// 128 threads (4 waves). Macro tile 64x128, BK=64, wave tile 32x64.
// Double-buffered LDS; staging is async global->LDS when available,
// else register-staged global loads overlapped with WMMA.
// OUT_MODE: 0 = bf16 row-major, 1 = f32 row-major, 2 = bf16 transposed.
// ---------------------------------------------------------------------------
template <int OUT_MODE>
__global__ __launch_bounds__(128) void gemm_bf16(const __bf16* __restrict__ A,
                                                 const __bf16* __restrict__ Bt,
                                                 void* __restrict__ Cv,
                                                 int M, int N, int K) {
    __shared__ __align__(16) __bf16 As[2][64][64];    // [buf][m][k]
    __shared__ __align__(16) __bf16 Bs[2][128][64];   // [buf][n][k]

    const int tid  = threadIdx.x;
    const int wave = tid >> 5;
    const int lane = tid & 31;
    const int m0   = blockIdx.y * 64;
    const int n0   = blockIdx.x * 128;
    const int wm   = (wave >> 1) * 32;
    const int wn   = (wave & 1) * 64;
    const int fn   = lane & 15;
    const int lg   = lane >> 4;

    const int ar_r  = tid >> 1;
    const int ar_cb = (tid & 1) * 32;

    v8f acc[2][4] = {};

    auto compute = [&](int cur) {
#pragma unroll
        for (int kc = 0; kc < 2; ++kc) {
            v16bf af[2], bfm[4];
#pragma unroll
            for (int mi = 0; mi < 2; ++mi) {
                v16bf f;
#pragma unroll
                for (int j = 0; j < 16; ++j) {
                    int kk = lg * 16 + j;
                    f[j] = As[cur][wm + mi * 16 + fn][kc * 32 + kk];
                }
                af[mi] = f;
            }
#pragma unroll
            for (int ni = 0; ni < 4; ++ni) {
                v16bf f;
#pragma unroll
                for (int j = 0; j < 16; ++j)
                    f[j] = Bs[cur][wn + ni * 16 + fn][kc * 32 + lg * 16 + j];
                bfm[ni] = f;
            }
#pragma unroll
            for (int mi = 0; mi < 2; ++mi)
#pragma unroll
                for (int ni = 0; ni < 4; ++ni)
                    acc[mi][ni] = __builtin_amdgcn_wmma_f32_16x16x32_bf16(
                        false, af[mi], false, bfm[ni], (short)0, acc[mi][ni], false, false);
        }
    };

    const int nk = K >> 6;

#if HAS_ASYNC_LDS
    auto stage_async = [&](int buf, int k0) {
        const __bf16* asrc = A + (size_t)(m0 + ar_r) * K + k0 + ar_cb;
#pragma unroll
        for (int u = 0; u < 4; ++u)
            async_b128(asrc + u * 8, &As[buf][ar_r][ar_cb + u * 8]);
        const __bf16* bsrc = Bt + (size_t)(n0 + tid) * K + k0;
#pragma unroll
        for (int u = 0; u < 8; ++u)
            async_b128(bsrc + u * 8, &Bs[buf][tid][u * 8]);
    };

    stage_async(0, 0);
    wait_async0();
    __syncthreads();
    for (int it = 0; it < nk; ++it) {
        const int cur = it & 1;
        if (it + 1 < nk) stage_async(cur ^ 1, (it + 1) << 6);   // fills other buffer
        compute(cur);
        wait_async0();
        __syncthreads();
    }
#else
    uint4 areg[4], breg[8];
    auto gload = [&](int k0) {
        const __bf16* asrc = A + (size_t)(m0 + ar_r) * K + k0 + ar_cb;
#pragma unroll
        for (int u = 0; u < 4; ++u) areg[u] = ((const uint4*)asrc)[u];
        const __bf16* bsrc = Bt + (size_t)(n0 + tid) * K + k0;
#pragma unroll
        for (int u = 0; u < 8; ++u) breg[u] = ((const uint4*)bsrc)[u];
        if (k0 + 64 < K) {
            __builtin_prefetch(asrc + 64, 0, 1);
            __builtin_prefetch(bsrc + 64, 0, 1);
        }
    };
    auto sstore = [&](int buf) {
#pragma unroll
        for (int u = 0; u < 4; ++u) *(uint4*)&As[buf][ar_r][ar_cb + u * 8] = areg[u];
#pragma unroll
        for (int u = 0; u < 8; ++u) *(uint4*)&Bs[buf][tid][u * 8] = breg[u];
    };

    gload(0);
    sstore(0);
    __syncthreads();
    for (int it = 0; it < nk; ++it) {
        const int cur = it & 1;
        if (it + 1 < nk) gload((it + 1) << 6);
        compute(cur);
        __syncthreads();
        if (it + 1 < nk) {
            sstore(cur ^ 1);
            __syncthreads();
        }
    }
#endif

    // ---- store (C layout: m = lg*8 + i, n = lane&15) ----
#pragma unroll
    for (int mi = 0; mi < 2; ++mi)
#pragma unroll
        for (int ni = 0; ni < 4; ++ni)
#pragma unroll
            for (int i = 0; i < 8; ++i) {
                size_t row = (size_t)(m0 + wm + mi * 16 + lg * 8 + i);
                size_t col = (size_t)(n0 + wn + ni * 16 + fn);
                if (OUT_MODE == 1)
                    ((float*)Cv)[row * N + col] = acc[mi][ni][i];
                else if (OUT_MODE == 2)
                    ((__bf16*)Cv)[col * M + row] = (__bf16)acc[mi][ni][i];
                else
                    ((__bf16*)Cv)[row * N + col] = (__bf16)acc[mi][ni][i];
            }
}

// ---------------------------------------------------------------------------
// Flash attention, causal. Block = 128 threads (4 waves), 64 queries/block.
// q: [B][T][16][128] bf16    k: [B][T][4][128] bf16
// vt: [4*128][B*T] bf16 (d-major, transposed)   out: [B][T][16][128] bf16
// K/V double-buffered in LDS; async global->LDS staging when available.
// ---------------------------------------------------------------------------
__global__ __launch_bounds__(128) void fattn(const __bf16* __restrict__ qg,
                                             const __bf16* __restrict__ kg,
                                             const __bf16* __restrict__ vt,
                                             __bf16* __restrict__ og) {
    __shared__ __align__(16) __bf16 Qs[64][128];
    __shared__ __align__(16) __bf16 Ks[2][64][128];    // [buf][key][d]
    __shared__ __align__(16) __bf16 Vts[2][128][64];   // [buf][d][key]
    __shared__ __align__(16) __bf16 Ps[4][16][64];

    const int tid  = threadIdx.x;
    const int wave = tid >> 5;
    const int lane = tid & 31;
    const int fn   = lane & 15;
    const int lg   = lane >> 4;

    const int qt = blockIdx.x;
    const int h  = blockIdx.y;
    const int b  = blockIdx.z;
    const int g  = h >> 2;
    const int q0 = qt * 64;

    const int kr_r  = tid >> 1;
    const int kr_cb = (tid & 1) * 64;

    float rowM[8], rowL[8];
    v8f o[8] = {};
#pragma unroll
    for (int i = 0; i < 8; ++i) { rowM[i] = -1e30f; rowL[i] = 0.0f; }

    const float scale = 0.08838834764831845f;   // 1/sqrt(128)
    const int nkt = qt + 1;                     // causal
    v16bf qf[4];

    auto compute_tile = [&](int cur, int kt) {
        // ---- scores S (16 x 64), contract over D = 128 ----
        v8f s[4] = {};
#pragma unroll
        for (int nt = 0; nt < 4; ++nt) {
#pragma unroll
            for (int kc = 0; kc < 4; ++kc) {
                v16bf bfK;      // B[kk][n] = Ks[nt*16+n][kc*32+kk] (k-contiguous)
#pragma unroll
                for (int j = 0; j < 16; ++j)
                    bfK[j] = Ks[cur][nt * 16 + fn][kc * 32 + lg * 16 + j];
                s[nt] = __builtin_amdgcn_wmma_f32_16x16x32_bf16(
                    false, qf[kc], false, bfK, (short)0, s[nt], false, false);
            }
        }

        // ---- scale + causal mask ----
#pragma unroll
        for (int nt = 0; nt < 4; ++nt)
#pragma unroll
            for (int i = 0; i < 8; ++i) {
                float val = s[nt][i] * scale;
                int qi = q0 + wave * 16 + lg * 8 + i;
                int kj = kt * 64 + nt * 16 + fn;
                if (kj > qi) val = -1e9f;
                s[nt][i] = val;
            }

        // ---- online softmax ----
#pragma unroll
        for (int i = 0; i < 8; ++i) {
            float mx = s[0][i];
#pragma unroll
            for (int nt = 1; nt < 4; ++nt) mx = fmaxf(mx, s[nt][i]);
#pragma unroll
            for (int off = 8; off >= 1; off >>= 1)
                mx = fmaxf(mx, __shfl_xor(mx, off, 16));
            float newM  = fmaxf(rowM[i], mx);
            float alpha = __expf(rowM[i] - newM);
            rowM[i] = newM;
            rowL[i] *= alpha;
#pragma unroll
            for (int d8 = 0; d8 < 8; ++d8) o[d8][i] *= alpha;

            float sm = 0.0f;
#pragma unroll
            for (int nt = 0; nt < 4; ++nt) {
                float p = __expf(s[nt][i] - newM);
                s[nt][i] = p;
                sm += p;
            }
#pragma unroll
            for (int off = 8; off >= 1; off >>= 1)
                sm += __shfl_xor(sm, off, 16);
            rowL[i] += sm;
        }

        // ---- P (C-layout) -> wave-private LDS -> A-layout ----
#pragma unroll
        for (int nt = 0; nt < 4; ++nt)
#pragma unroll
            for (int i = 0; i < 8; ++i)
                Ps[wave][lg * 8 + i][nt * 16 + fn] = (__bf16)s[nt][i];

        // ---- O += P @ V (contract over 64 keys, N = D = 128) ----
#pragma unroll
        for (int kc = 0; kc < 2; ++kc) {
            v16bf pf;
#pragma unroll
            for (int j = 0; j < 16; ++j) {
                int kk = lg * 16 + j;
                pf[j] = Ps[wave][fn][kc * 32 + kk];
            }
#pragma unroll
            for (int d8 = 0; d8 < 8; ++d8) {
                v16bf vf;   // B[key][d] from Vts[d][key]: key-contiguous
#pragma unroll
                for (int j = 0; j < 16; ++j)
                    vf[j] = Vts[cur][d8 * 16 + fn][kc * 32 + lg * 16 + j];
                o[d8] = __builtin_amdgcn_wmma_f32_16x16x32_bf16(
                    false, pf, false, vf, (short)0, o[d8], false, false);
            }
        }
    };

#if HAS_ASYNC_LDS
    auto stage_kv_async = [&](int buf, int kt) {
        const int t0 = kt * 64;
        const __bf16* ks = kg + (((size_t)b * T_DIM + t0 + kr_r) * G_DIM + g) * D_DIM + kr_cb;
#pragma unroll
        for (int u = 0; u < 4; ++u)
            async_b128(ks + u * 8, &Ks[buf][kr_r][kr_cb + u * 8]);
        const __bf16* vs = vt + ((size_t)g * D_DIM + tid) * NTOK + (size_t)b * T_DIM + t0;
#pragma unroll
        for (int u = 0; u < 8; ++u)
            async_b128(vs + u * 8, &Vts[buf][tid][u * 8]);
    };

    stage_kv_async(0, 0);        // K/V tile 0 in flight (ASYNCcnt)
#else
    uint4 kreg[8], vreg[8];
    auto gload_kv = [&](int kt) {
        const int t0 = kt * 64;
        const __bf16* ks = kg + (((size_t)b * T_DIM + t0 + kr_r) * G_DIM + g) * D_DIM + kr_cb;
#pragma unroll
        for (int u = 0; u < 8; ++u) kreg[u] = *(const uint4*)(ks + u * 8);
        const __bf16* vs = vt + ((size_t)g * D_DIM + tid) * NTOK + (size_t)b * T_DIM + t0;
#pragma unroll
        for (int u = 0; u < 8; ++u) vreg[u] = ((const uint4*)vs)[u];
    };
    auto sstore_kv = [&](int buf) {
#pragma unroll
        for (int u = 0; u < 8; ++u) *(uint4*)&Ks[buf][kr_r][kr_cb + u * 8] = kreg[u];
#pragma unroll
        for (int u = 0; u < 8; ++u) *(uint4*)&Vts[buf][tid][u * 8] = vreg[u];
    };

    gload_kv(0);                 // K/V tile 0 in flight
#endif

    // ---- stage Q tile: 64 x 128 (overlapped with K/V tile 0) ----
    {
        const int r  = tid >> 1;
        const int cb = (tid & 1) * 64;
        const __bf16* src = qg + (((size_t)b * T_DIM + q0 + r) * HQ_DIM + h) * D_DIM + cb;
#pragma unroll
        for (int u = 0; u < 8; ++u)
            *(uint4*)&Qs[r][cb + u * 8] = *(const uint4*)(src + u * 8);
    }

#if HAS_ASYNC_LDS
    wait_async0();
    __syncthreads();
#else
    sstore_kv(0);
    __syncthreads();
#endif

    // ---- q fragments: 4 A-frags per wave, kept in registers ----
#pragma unroll
    for (int kc = 0; kc < 4; ++kc) {
        v16bf f;
#pragma unroll
        for (int j = 0; j < 16; ++j) {
            int kk = lg * 16 + j;
            f[j] = Qs[wave * 16 + fn][kc * 32 + kk];
        }
        qf[kc] = f;
    }

#if HAS_ASYNC_LDS
    for (int kt = 0; kt < nkt; ++kt) {
        const int cur = kt & 1;
        if (kt + 1 < nkt) stage_kv_async(cur ^ 1, kt + 1);
        compute_tile(cur, kt);
        wait_async0();
        __syncthreads();
    }
#else
    for (int kt = 0; kt < nkt; ++kt) {
        const int cur = kt & 1;
        if (kt + 1 < nkt) gload_kv(kt + 1);
        compute_tile(cur, kt);
        __syncthreads();
        if (kt + 1 < nkt) {
            sstore_kv(cur ^ 1);
            __syncthreads();
        }
    }
#endif

    // ---- normalize and store O (bf16, [B][T][16][128]) ----
#pragma unroll
    for (int d8 = 0; d8 < 8; ++d8)
#pragma unroll
        for (int i = 0; i < 8; ++i) {
            int qi  = q0 + wave * 16 + lg * 8 + i;
            int col = d8 * 16 + fn;
            float val = o[d8][i] / rowL[i];
            og[(((size_t)b * T_DIM + qi) * HQ_DIM + h) * D_DIM + col] = (__bf16)val;
        }
}

// ---------------------------------------------------------------------------
// Host-side orchestration
// ---------------------------------------------------------------------------
extern "C" void kernel_launch(void* const* d_in, const int* in_sizes, int n_in,
                              void* d_out, int out_size, void* d_ws, size_t ws_size,
                              hipStream_t stream) {
    (void)in_sizes; (void)n_in; (void)out_size; (void)ws_size;
    const float* x  = (const float*)d_in[0];
    const float* Wq = (const float*)d_in[1];
    const float* Wk = (const float*)d_in[2];
    const float* Wv = (const float*)d_in[3];
    const float* Wo = (const float*)d_in[4];

    const size_t NQ  = (size_t)HQ_DIM * D_DIM;   // 2048
    const size_t NKV = (size_t)G_DIM * D_DIM;    // 512

    char* ws = (char*)d_ws;
    size_t off = 0;
    auto carve = [&](size_t bytes) {
        char* p = ws + off;
        off += (bytes + 255) & ~(size_t)255;
        return p;
    };
    __bf16* xb   = (__bf16*)carve((size_t)NTOK * E_DIM * 2);
    __bf16* Wqt  = (__bf16*)carve((size_t)E_DIM * NQ * 2);    // [NQ][E]
    __bf16* Wkt  = (__bf16*)carve((size_t)E_DIM * NKV * 2);   // [NKV][E]
    __bf16* Wvt  = (__bf16*)carve((size_t)E_DIM * NKV * 2);   // [NKV][E]
    __bf16* Wot  = (__bf16*)carve(NQ * (size_t)E_DIM * 2);    // [E][NQ]
    __bf16* qb   = (__bf16*)carve((size_t)NTOK * NQ * 2);     // [B][T][16][128]
    __bf16* kb   = (__bf16*)carve((size_t)NTOK * NKV * 2);    // [B][T][4][128]
    __bf16* vtb  = (__bf16*)carve((size_t)NTOK * NKV * 2);    // [512][8192]
    __bf16* ab   = (__bf16*)carve((size_t)NTOK * NQ * 2);     // [B][T][16][128]

    {
        int n4 = (int)((size_t)NTOK * E_DIM / 4);
        cvt_f32_bf16<<<(n4 + 255) / 256, 256, 0, stream>>>(x, xb, n4);
    }
    cvt_transpose_bf16<<<dim3((int)(NQ / 64),  E_DIM / 64), 256, 0, stream>>>(Wq, Wqt, E_DIM, (int)NQ);
    cvt_transpose_bf16<<<dim3((int)(NKV / 64), E_DIM / 64), 256, 0, stream>>>(Wk, Wkt, E_DIM, (int)NKV);
    cvt_transpose_bf16<<<dim3((int)(NKV / 64), E_DIM / 64), 256, 0, stream>>>(Wv, Wvt, E_DIM, (int)NKV);
    cvt_transpose_bf16<<<dim3(E_DIM / 64, (int)(NQ / 64)),  256, 0, stream>>>(Wo, Wot, (int)NQ, E_DIM);

    dim3 blk(128);
    gemm_bf16<0><<<dim3((int)(NQ / 128),  NTOK / 64), blk, 0, stream>>>(xb, Wqt, qb,  NTOK, (int)NQ,  E_DIM);
    gemm_bf16<0><<<dim3((int)(NKV / 128), NTOK / 64), blk, 0, stream>>>(xb, Wkt, kb,  NTOK, (int)NKV, E_DIM);
    gemm_bf16<2><<<dim3((int)(NKV / 128), NTOK / 64), blk, 0, stream>>>(xb, Wvt, vtb, NTOK, (int)NKV, E_DIM);

    rope_kernel<<<(int)((size_t)NTOK * HQ_DIM * 64 / 256), 256, 0, stream>>>(qb, HQ_DIM);
    rope_kernel<<<(int)((size_t)NTOK * G_DIM  * 64 / 256), 256, 0, stream>>>(kb, G_DIM);

    fattn<<<dim3(T_DIM / 64, HQ_DIM, B_DIM), blk, 0, stream>>>(qb, kb, vtb, ab);

    gemm_bf16<1><<<dim3(E_DIM / 128, NTOK / 64), blk, 0, stream>>>(ab, Wot, (float*)d_out, NTOK, E_DIM, (int)NQ);
}
